// Block_50929722196345
// MI455X (gfx1250) — matrix operand order repaired and checked by reading, compiled, not run on hardware
//
#include <hip/hip_runtime.h>
#include <hip/hip_bf16.h>

// ---------------------------------------------------------------------------
// Transformer block forward for MI455X (gfx1250), bf16 WMMA + fp32 accumulate.
//
// Roofline: ~138 GFLOP, ~150MB unique bytes (resident in 192MB L2) -> compute
// bound; all matmuls run through v_wmma_f32_16x16x32_bf16. Flash-style
// attention (no SxS materialization) with the causal mask hoisted out of the
// interior key tiles (mask VALU only on the 1-2 diagonal tiles). Weights are
// pre-transposed to [N,K] bf16 so B-fragments are contiguous 32B loads; V is
// stored per-head transposed so P@V B-fragments are contiguous too.
// Workspace requirement: 128 MB.
// ---------------------------------------------------------------------------

typedef __bf16 bf16_t;
typedef __bf16 v16bf __attribute__((ext_vector_type(16)));
typedef __bf16 v8bf  __attribute__((ext_vector_type(8)));
typedef float  v8f   __attribute__((ext_vector_type(8)));

#define BATCH     2
#define SEQ       2048
#define DMODEL    1024
#define NHEADS    16
#define HDIM      64
#define DFF       4096
#define NTOK      (BATCH * SEQ)
#define ATT_SCALE 0.125f     // 1/sqrt(64)
#define LN_EPS    1e-6f
#define NEG_BIG   (-1e30f)

union V16 { v16bf v; v8bf h[2]; };

static __device__ __forceinline__ v8f wmma_bf16(v16bf a, v16bf b, v8f c) {
  // D = A(16x32 bf16) * B(32x16 bf16) + C(16x16 f32)
  return __builtin_amdgcn_wmma_f32_16x16x32_bf16(false, a, false, b,
                                                 (short)0, c, false, false);
}

// ---------------------------------------------------------------------------
// Elementwise f32 -> bf16
// ---------------------------------------------------------------------------
__global__ void cvt_f32_bf16(const float* __restrict__ in,
                             bf16_t* __restrict__ out, int n) {
  int i = blockIdx.x * blockDim.x + threadIdx.x;
  if (i < n) out[i] = (bf16_t)in[i];
}

// ---------------------------------------------------------------------------
// Weight transpose + convert: w[K,N] f32 (row-major) -> wt[N,K] bf16.
// Coalesced writes along K.
// ---------------------------------------------------------------------------
__global__ void transpose_cvt(const float* __restrict__ w,
                              bf16_t* __restrict__ wt, int K, int N) {
  long i = (long)blockIdx.x * blockDim.x + threadIdx.x;
  long total = (long)K * N;
  if (i >= total) return;
  int k = (int)(i % K);
  int n = (int)(i / K);
  wt[(long)n * K + k] = (bf16_t)w[(long)k * N + n];
}

// ---------------------------------------------------------------------------
// Generic WMMA GEMM: C[M,N] = A[M,K](bf16,row) * Bt[N,K](bf16,row) + bias.
// Per-wave tile: 32(M) x 64(N): 8 WMMAs per K=32 step, B-fragments reused
// across both M-fragments. Epilogue selected by EPI.
// ---------------------------------------------------------------------------
enum { EP_BF16_ROW = 0, EP_VT = 1, EP_F32_RES = 2, EP_BF16_GELU = 3 };

template <int EPI>
__global__ __launch_bounds__(256) void gemm_wmma(
    const bf16_t* __restrict__ A, const bf16_t* __restrict__ Bt,
    const float* __restrict__ bias, const float* __restrict__ res,
    void* __restrict__ out, int Mdim, int Ndim, int Kdim) {
  const int lane  = threadIdx.x & 31;
  const int wave  = (blockIdx.x * blockDim.x + threadIdx.x) >> 5;
  const int ntile = Ndim >> 6;
  const int m0    = (wave / ntile) * 32;
  const int n0    = (wave % ntile) * 64;
  const int lm    = lane & 15;   // row (A: M%16) or column (B/C: N%16)
  const int lh    = lane >> 4;   // half selector

  v8f c[2][4];
#pragma unroll
  for (int i = 0; i < 2; i++)
#pragma unroll
    for (int j = 0; j < 4; j++)
#pragma unroll
      for (int v = 0; v < 8; v++) c[i][j][v] = 0.0f;

  const bf16_t* arow0 = A + (long)(m0 + lm) * Kdim;
  const bf16_t* arow1 = A + (long)(m0 + 16 + lm) * Kdim;

  for (int kt = 0; kt < Kdim; kt += 32) {
    V16 a0, a1;
    // A fragment 16x32: lane m holds K = kt + 16h .. kt + 16h + 15 (contig)
    a0.h[0] = *(const v8bf*)(arow0 + kt + 16 * lh);
    a0.h[1] = *(const v8bf*)(arow0 + kt + 16 * lh + 8);
    a1.h[0] = *(const v8bf*)(arow1 + kt + 16 * lh);
    a1.h[1] = *(const v8bf*)(arow1 + kt + 16 * lh + 8);
#pragma unroll
    for (int j = 0; j < 4; j++) {
      // B fragment 32x16: lane n holds K = kt + 16h .. kt + 16h + 15 (contig)
      v16bf b = *(const v16bf*)(Bt + (long)(n0 + j * 16 + lm) * Kdim + kt + 16 * lh);
      c[0][j] = wmma_bf16(a0.v, b, c[0][j]);
      c[1][j] = wmma_bf16(a1.v, b, c[1][j]);
    }
  }

#pragma unroll
  for (int mf = 0; mf < 2; mf++) {
#pragma unroll
    for (int nf = 0; nf < 4; nf++) {
      const int col = n0 + nf * 16 + lm;
      const float bv = bias[col];
      if (EPI == EP_BF16_ROW) {
        bf16_t* o = (bf16_t*)out;
#pragma unroll
        for (int v = 0; v < 8; v++) {
          int row = m0 + mf * 16 + v + 8 * lh;
          o[(long)row * Ndim + col] = (bf16_t)(c[mf][nf][v] + bv);
        }
      } else if (EPI == EP_F32_RES) {
        float* o = (float*)out;
#pragma unroll
        for (int v = 0; v < 8; v++) {
          int row = m0 + mf * 16 + v + 8 * lh;
          long idx = (long)row * Ndim + col;
          o[idx] = res[idx] + c[mf][nf][v] + bv;
        }
      } else if (EPI == EP_BF16_GELU) {
        bf16_t* o = (bf16_t*)out;
#pragma unroll
        for (int v = 0; v < 8; v++) {
          int row = m0 + mf * 16 + v + 8 * lh;
          float x = c[mf][nf][v] + bv;
          float g = 0.5f * x *
                    (1.0f + tanhf(0.7978845608f * (x + 0.044715f * x * x * x)));
          o[(long)row * Ndim + col] = (bf16_t)g;
        }
      } else {  // EP_VT : write Vt[b][head][d][s], 8 consecutive tokens packed
        bf16_t* o = (bf16_t*)out;
        int head = col >> 6, d = col & 63;
        int t0 = m0 + mf * 16 + 8 * lh;  // rows v=0..7 are consecutive tokens
        int bb = t0 / SEQ, s0 = t0 % SEQ;
        v8bf pack;
#pragma unroll
        for (int v = 0; v < 8; v++) pack[v] = (bf16_t)(c[mf][nf][v] + bv);
        *(v8bf*)(o + ((long)(bb * NHEADS + head) * HDIM + d) * SEQ + s0) = pack;
      }
    }
  }
}

// ---------------------------------------------------------------------------
// One 16(q) x 32(k) flash-attention tile. MASKED=true only for the 1-2
// diagonal tiles; interior tiles skip all causal compares. Padding-mask
// cndmasks are guarded by a wave-uniform ballot (mask is usually all-ones).
// ---------------------------------------------------------------------------
template <bool MASKED>
static __device__ __forceinline__ void attn_tile(
    int kt, int qbase, int lm, int lh, const V16* aQ, const bf16_t* Kbase,
    const bf16_t* Vbase, const int* __restrict__ mrow, bf16_t* sPw, v8f* o,
    float* l, float* mx) {
  // ---- scores S = Q K^T (16 x 32), two C fragments (key halves) ----
  v8f s0, s1;
#pragma unroll
  for (int v = 0; v < 8; v++) { s0[v] = 0.0f; s1[v] = 0.0f; }
#pragma unroll
  for (int cc = 0; cc < 2; cc++) {
    const bf16_t* kp = Kbase + (long)(kt * 32) * DMODEL + cc * 32 + 16 * lh;
    v16bf bk0 = *(const v16bf*)(kp + (long)lm * DMODEL);
    v16bf bk1 = *(const v16bf*)(kp + (long)(lm + 16) * DMODEL);
    s0 = wmma_bf16(aQ[cc].v, bk0, s0);
    s1 = wmma_bf16(aQ[cc].v, bk1, s1);
  }

  const int key0 = kt * 32 + lm;
  const int key1 = key0 + 16;
  const bool pm0 = mrow[key0] != 0;
  const bool pm1 = mrow[key1] != 0;
  // wave-uniform: skip padding cndmasks when the whole tile is unpadded
  const bool padded = ((unsigned)__ballot(pm0 && pm1) != 0xFFFFFFFFu);

  float p0s[8], p1s[8];
#pragma unroll
  for (int v = 0; v < 8; v++) {
    float x0 = s0[v] * ATT_SCALE;
    float x1 = s1[v] * ATT_SCALE;
    if (MASKED) {
      const int row = qbase + v + 8 * lh;
      if (key0 > row) x0 = NEG_BIG;
      if (key1 > row) x1 = NEG_BIG;
    }
    if (padded) {
      if (!pm0) x0 = NEG_BIG;
      if (!pm1) x1 = NEG_BIG;
    }
    float tmax = fmaxf(x0, x1);
#pragma unroll
    for (int d = 1; d < 16; d <<= 1) tmax = fmaxf(tmax, __shfl_xor(tmax, d, 32));
    float nm   = fmaxf(mx[v], tmax);
    float corr = __expf(mx[v] - nm);
    mx[v] = nm;
    float e0 = __expf(x0 - nm), e1 = __expf(x1 - nm);
    float rs = e0 + e1;
#pragma unroll
    for (int d = 1; d < 16; d <<= 1) rs += __shfl_xor(rs, d, 32);
    l[v] = l[v] * corr + rs;
#pragma unroll
    for (int f = 0; f < 4; f++) o[f][v] *= corr;
    p0s[v] = e0;
    p1s[v] = e1;
  }

  // ---- P: C-layout -> LDS -> A-layout (per-wave region, in-order DS) ----
  asm volatile("" ::: "memory");
#pragma unroll
  for (int v = 0; v < 8; v++) {
    int r = v + 8 * lh;
    sPw[r * 32 + lm]      = (bf16_t)p0s[v];
    sPw[r * 32 + 16 + lm] = (bf16_t)p1s[v];
  }
  asm volatile("s_wait_dscnt 0" ::: "memory");
  V16 pA;
  pA.h[0] = *(const v8bf*)(&sPw[lm * 32 + 16 * lh]);
  pA.h[1] = *(const v8bf*)(&sPw[lm * 32 + 16 * lh + 8]);
  asm volatile("" ::: "memory");

  // ---- O += P (16x32) @ V (32x64); Vt gives contiguous key-pairs ----
  const bf16_t* vp = Vbase + kt * 32 + 16 * lh;
#pragma unroll
  for (int f = 0; f < 4; f++) {
    v16bf bv = *(const v16bf*)(vp + (long)(f * 16 + lm) * SEQ);
    o[f] = wmma_bf16(pA.v, bv, o[f]);
  }
}

// ---------------------------------------------------------------------------
// Flash attention: one wave = one (b, head, 16-row q tile). Streams 32-key
// tiles with online softmax; causal mask applied only on diagonal tiles.
// ---------------------------------------------------------------------------
__global__ __launch_bounds__(256) void flash_attn(
    const bf16_t* __restrict__ Q, const bf16_t* __restrict__ Kmat,
    const bf16_t* __restrict__ Vt, const int* __restrict__ amask,
    bf16_t* __restrict__ O) {
  __shared__ __align__(32) bf16_t sP[8][16 * 32];  // per-wave 16x32 P tile
  const int lane  = threadIdx.x & 31;
  const int wslot = threadIdx.x >> 5;
  const int wave  = (blockIdx.x * blockDim.x + threadIdx.x) >> 5;
  const int qt    = wave & 127;           // SEQ/16 = 128 q tiles
  const int head  = (wave >> 7) & (NHEADS - 1);
  const int b     = wave >> 11;
  const int lm    = lane & 15;
  const int lh    = lane >> 4;
  const int qbase = qt * 16;

  const bf16_t* Kbase = Kmat + (long)b * SEQ * DMODEL + head * HDIM;
  const bf16_t* Vbase = Vt + (long)(b * NHEADS + head) * HDIM * SEQ;
  const int*    mrow  = amask + b * SEQ;

  // Q A-fragments for head-dim chunks 0..31 and 32..63 (kept in registers)
  V16 aQ[2];
  {
    const bf16_t* qrow = Q + (long)(b * SEQ + qbase + lm) * DMODEL + head * HDIM;
#pragma unroll
    for (int cc = 0; cc < 2; cc++) {
      aQ[cc].h[0] = *(const v8bf*)(qrow + cc * 32 + 16 * lh);
      aQ[cc].h[1] = *(const v8bf*)(qrow + cc * 32 + 16 * lh + 8);
    }
  }

  v8f o[4];
#pragma unroll
  for (int f = 0; f < 4; f++)
#pragma unroll
    for (int v = 0; v < 8; v++) o[f][v] = 0.0f;
  float l[8], mx[8];
#pragma unroll
  for (int v = 0; v < 8; v++) { l[v] = 0.0f; mx[v] = NEG_BIG; }

  const int ktfull = qbase / 32;            // tiles fully below the diagonal
  const int ktmax  = (qbase + 15) / 32 + 1; // causal: skip tiles past diagonal
  for (int kt = 0; kt < ktfull; kt++)
    attn_tile<false>(kt, qbase, lm, lh, aQ, Kbase, Vbase, mrow, sP[wslot], o, l, mx);
  for (int kt = ktfull; kt < ktmax; kt++)
    attn_tile<true>(kt, qbase, lm, lh, aQ, Kbase, Vbase, mrow, sP[wslot], o, l, mx);

  // ---- normalize and store attn output (bf16, row-major [tok, D]) ----
#pragma unroll
  for (int f = 0; f < 4; f++) {
#pragma unroll
    for (int v = 0; v < 8; v++) {
      int row = qbase + v + 8 * lh;
      float val = o[f][v] / l[v];
      O[(long)(b * SEQ + row) * DMODEL + head * HDIM + f * 16 + lm] = (bf16_t)val;
    }
  }
}

// ---------------------------------------------------------------------------
// LayerNorm over last dim (1024): one block per row. Writes f32 and
// optionally a bf16 copy (A operand of the following GEMM).
// ---------------------------------------------------------------------------
__global__ __launch_bounds__(256) void layernorm_k(
    const float* __restrict__ in, const float* __restrict__ gamma,
    const float* __restrict__ beta, float* __restrict__ outf,
    bf16_t* __restrict__ outb) {
  __shared__ float red[16];
  const long base = (long)blockIdx.x * DMODEL;
  float s = 0.f, s2 = 0.f;
  for (int i = threadIdx.x; i < DMODEL; i += 256) {
    float v = in[base + i];
    s += v;
    s2 += v * v;
  }
#pragma unroll
  for (int d = 1; d < 32; d <<= 1) {
    s += __shfl_xor(s, d, 32);
    s2 += __shfl_xor(s2, d, 32);
  }
  int wid = threadIdx.x >> 5;
  if ((threadIdx.x & 31) == 0) { red[wid] = s; red[wid + 8] = s2; }
  __syncthreads();
  if (threadIdx.x < 32) {
    float a = (threadIdx.x < 8) ? red[threadIdx.x] : 0.f;
    float c = (threadIdx.x < 8) ? red[threadIdx.x + 8] : 0.f;
#pragma unroll
    for (int d = 1; d < 8; d <<= 1) {
      a += __shfl_xor(a, d, 32);
      c += __shfl_xor(c, d, 32);
    }
    if (threadIdx.x == 0) { red[0] = a; red[1] = c; }
  }
  __syncthreads();
  float mean = red[0] * (1.0f / DMODEL);
  float var  = red[1] * (1.0f / DMODEL) - mean * mean;
  float inv  = rsqrtf(var + LN_EPS);
  for (int i = threadIdx.x; i < DMODEL; i += 256) {
    float v = (in[base + i] - mean) * inv * gamma[i] + beta[i];
    outf[base + i] = v;
    if (outb) outb[base + i] = (bf16_t)v;
  }
}

// ---------------------------------------------------------------------------
// Host-side orchestration
// ---------------------------------------------------------------------------
extern "C" void kernel_launch(void* const* d_in, const int* in_sizes, int n_in,
                              void* d_out, int out_size, void* d_ws,
                              size_t ws_size, hipStream_t stream) {
  (void)in_sizes; (void)n_in; (void)out_size; (void)ws_size;
  const float* x     = (const float*)d_in[0];
  const int*   amask = (const int*)d_in[1];
  const float* wq = (const float*)d_in[2];  const float* bq = (const float*)d_in[3];
  const float* wk = (const float*)d_in[4];  const float* bk = (const float*)d_in[5];
  const float* wv = (const float*)d_in[6];  const float* bv = (const float*)d_in[7];
  const float* wo = (const float*)d_in[8];  const float* bo = (const float*)d_in[9];
  const float* g1 = (const float*)d_in[10]; const float* be1 = (const float*)d_in[11];
  const float* w1 = (const float*)d_in[12]; const float* b1 = (const float*)d_in[13];
  const float* w2 = (const float*)d_in[14]; const float* b2 = (const float*)d_in[15];
  const float* g2 = (const float*)d_in[16]; const float* be2 = (const float*)d_in[17];

  char* ws = (char*)d_ws;
  size_t off = 0;
  auto alloc = [&](size_t bytes) -> void* {
    void* p = ws + off;
    off += (bytes + 255) & ~(size_t)255;
    return p;
  };
  bf16_t* xb    = (bf16_t*)alloc((size_t)NTOK * DMODEL * 2);   // 8 MB
  bf16_t* wqt   = (bf16_t*)alloc((size_t)DMODEL * DMODEL * 2); // 2 MB
  bf16_t* wkt   = (bf16_t*)alloc((size_t)DMODEL * DMODEL * 2);
  bf16_t* wvt   = (bf16_t*)alloc((size_t)DMODEL * DMODEL * 2);
  bf16_t* wot   = (bf16_t*)alloc((size_t)DMODEL * DMODEL * 2);
  bf16_t* w1t   = (bf16_t*)alloc((size_t)DFF * DMODEL * 2);    // 8 MB
  bf16_t* w2t   = (bf16_t*)alloc((size_t)DMODEL * DFF * 2);    // 8 MB
  bf16_t* Qb    = (bf16_t*)alloc((size_t)NTOK * DMODEL * 2);   // 8 MB
  bf16_t* Kb    = (bf16_t*)alloc((size_t)NTOK * DMODEL * 2);
  bf16_t* Vt    = (bf16_t*)alloc((size_t)NTOK * DMODEL * 2);
  bf16_t* attnb = (bf16_t*)alloc((size_t)NTOK * DMODEL * 2);
  float*  res1  = (float*)alloc((size_t)NTOK * DMODEL * 4);    // 16 MB
  float*  x1f   = (float*)alloc((size_t)NTOK * DMODEL * 4);    // 16 MB
  bf16_t* hb    = (bf16_t*)alloc((size_t)NTOK * DFF * 2);      // 32 MB
  bf16_t* x1b   = Qb;   // reuse (Q dead after attention)
  float*  y     = res1; // reuse (res1 dead after LN1)

  const int TPB = 256;
  // 0) activations -> bf16
  cvt_f32_bf16<<<(NTOK * DMODEL + TPB - 1) / TPB, TPB, 0, stream>>>(
      x, xb, NTOK * DMODEL);
  // 1) weight transpose + convert ([K,N] f32 -> [N,K] bf16)
  transpose_cvt<<<(DMODEL * DMODEL + TPB - 1) / TPB, TPB, 0, stream>>>(wq, wqt, DMODEL, DMODEL);
  transpose_cvt<<<(DMODEL * DMODEL + TPB - 1) / TPB, TPB, 0, stream>>>(wk, wkt, DMODEL, DMODEL);
  transpose_cvt<<<(DMODEL * DMODEL + TPB - 1) / TPB, TPB, 0, stream>>>(wv, wvt, DMODEL, DMODEL);
  transpose_cvt<<<(DMODEL * DMODEL + TPB - 1) / TPB, TPB, 0, stream>>>(wo, wot, DMODEL, DMODEL);
  transpose_cvt<<<(DMODEL * DFF + TPB - 1) / TPB, TPB, 0, stream>>>(w1, w1t, DMODEL, DFF);
  transpose_cvt<<<(DFF * DMODEL + TPB - 1) / TPB, TPB, 0, stream>>>(w2, w2t, DFF, DMODEL);

  auto gemm_grid = [&](int M, int N) { return ((M / 32) * (N / 64) * 32) / TPB; };
  // 2) Q, K projections (bf16 row-major out), V projection (per-head transposed)
  gemm_wmma<EP_BF16_ROW><<<gemm_grid(NTOK, DMODEL), TPB, 0, stream>>>(
      xb, wqt, bq, nullptr, Qb, NTOK, DMODEL, DMODEL);
  gemm_wmma<EP_BF16_ROW><<<gemm_grid(NTOK, DMODEL), TPB, 0, stream>>>(
      xb, wkt, bk, nullptr, Kb, NTOK, DMODEL, DMODEL);
  gemm_wmma<EP_VT><<<gemm_grid(NTOK, DMODEL), TPB, 0, stream>>>(
      xb, wvt, bv, nullptr, Vt, NTOK, DMODEL, DMODEL);
  // 3) flash attention: B*H*(S/16) = 4096 waves
  flash_attn<<<(BATCH * NHEADS * (SEQ / 16) * 32) / TPB, TPB, 0, stream>>>(
      Qb, Kb, Vt, amask, attnb);
  // 4) output projection + residual (f32)
  gemm_wmma<EP_F32_RES><<<gemm_grid(NTOK, DMODEL), TPB, 0, stream>>>(
      attnb, wot, bo, x, res1, NTOK, DMODEL, DMODEL);
  // 5) LN1 -> x1 (f32 + bf16 copy)
  layernorm_k<<<NTOK, TPB, 0, stream>>>(res1, g1, be1, x1f, x1b);
  // 6) FFN up + GELU (bf16 out)
  gemm_wmma<EP_BF16_GELU><<<gemm_grid(NTOK, DFF), TPB, 0, stream>>>(
      x1b, w1t, b1, nullptr, hb, NTOK, DFF, DMODEL);
  // 7) FFN down + residual (f32)
  gemm_wmma<EP_F32_RES><<<gemm_grid(NTOK, DMODEL), TPB, 0, stream>>>(
      hb, w2t, b2, x1f, y, NTOK, DMODEL, DFF);
  // 8) LN2 -> d_out (f32)
  layernorm_k<<<NTOK, TPB, 0, stream>>>(y, g2, be2, (float*)d_out, nullptr);
}
